// EndoSurfRenderer_29308856828391
// MI455X (gfx1250) — compile-verified, matrix operand order checked by reading, not run on hardware
//
#include <hip/hip_runtime.h>
#include <hip/hip_bf16.h>
#include <math.h>

// ---------------------------------------------------------------------------
// EndoSurf renderer, fused WMMA implementation for gfx1250 (MI455X).
//  - f16 WMMA (v_wmma_f32_16x16x32_f16), f32 accumulate
//  - per block: 16 points, all activations in LDS
//  - deform JVP and SDF gradient computed as forward-mode tangent channels
//    sharing the same WMMA GEMMs as the primal pass
//  - weights pre-blocked to per-lane-contiguous B fragments (global_load_b128)
//  - register-blocked RTxCW tiles per wave: RT+CW fragment loads feed RT*CW
//    back-to-back WMMAs per K step, no divergent/scalar guards in the hot loop
// ---------------------------------------------------------------------------

typedef __attribute__((ext_vector_type(16))) _Float16 v16h;
typedef __attribute__((ext_vector_type(8)))  _Float16 v8h;
typedef __attribute__((ext_vector_type(8)))  float    v8f;

// explicit global-address-space pointers so weight/bias traffic lowers to
// global_load_* (GVS: SGPR base + VGPR offset) instead of flat_load_*
typedef const __attribute__((address_space(1))) _Float16 gh_t;
typedef const __attribute__((address_space(1))) v8h      gv8h_t;
typedef const __attribute__((address_space(1))) float    gf_t;

#define ST      352        // LDS activation row stride (halfs), max padded K
#define NLAYERS 23

struct LayerDesc { int offW, offB, ktiles, ctiles; };
struct AllDesc   { LayerDesc L[NLAYERS]; };

// ---------------------------------------------------------------------------
// fragment loads
// A (16x32 f16): lane<16 -> row=rt*16+lane, K runs {kb..kb+7, kb+16..kb+23};
//                lane>=16 -> same rows, K runs shifted by +8.  Two 16B LDS loads.
__device__ __forceinline__ v16h load_frag_a(const _Float16* p) {
    v8h lo = *(const v8h*)(p);
    v8h hi = *(const v8h*)(p + 16);
    return __builtin_shufflevector(lo, hi, 0,1,2,3,4,5,6,7,8,9,10,11,12,13,14,15);
}
// B (32x16 f16): pre-blocked so each lane's 16 halfs are contiguous (32B).
__device__ __forceinline__ v16h load_frag_b(gh_t* p) {
    v8h lo = *(gv8h_t*)(p);
    v8h hi = *(gv8h_t*)(p + 8);
    return __builtin_shufflevector(lo, hi, 0,1,2,3,4,5,6,7,8,9,10,11,12,13,14,15);
}

__device__ __forceinline__ float sigmoidf_(float z) { return 1.f / (1.f + __expf(-z)); }
__device__ __forceinline__ float softplus100(float x) {
    float z = 100.f * x;
    if (z > 15.f) return x;
    return log1pf(__expf(z)) * 0.01f;
}

// ---------------------------------------------------------------------------
// One dense layer:  acts[0..RT*16)[0..K) @ W + b -> acts[..][0..Npad) (pre-act),
// in place (register-staged accumulators + barriers).
// Wave `w` owns column tiles ct = w + 8j (j<CW), clamped to the last valid
// tile; clamped duplicates recompute identical data, so their stores are
// benign. No guards -> EXEC stays all-ones across every WMMA (ISA req.)
// and the K loop is pure loads + WMMA bursts.
template<int RT, int CW>
__device__ __forceinline__
void run_layer_t(_Float16* acts,
                 gh_t* __restrict__ wbuf, gf_t* __restrict__ bbuf,
                 LayerDesc L, int wave, int lane)
{
    int ctj [CW];   // owned column tiles (clamped)
    int wbase[CW];  // global half-offset of B fragment at kt=0
#pragma unroll
    for (int j = 0; j < CW; ++j) {
        int c = wave + (j << 3);
        if (c > L.ctiles - 1) c = L.ctiles - 1;
        ctj[j]   = c;
        wbase[j] = c * L.ktiles * 512 + lane * 16;
    }
    int abase[RT];  // LDS half-offset of A fragment at kt=0
#pragma unroll
    for (int r = 0; r < RT; ++r)
        abase[r] = ((r << 4) + (lane & 15)) * ST + ((lane >> 4) << 3);

    v8f acc[RT][CW] = {};
    gh_t* wb = wbuf + L.offW;

    for (int kt = 0; kt < L.ktiles; ++kt) {
        const int ka = kt * 32;     // A advances 32 halfs per K tile
        const int kb = kt * 512;    // B advances one 512-half fragment block
        v16h A[RT], B[CW];
#pragma unroll
        for (int r = 0; r < RT; ++r) A[r] = load_frag_a(acts + abase[r] + ka);
#pragma unroll
        for (int j = 0; j < CW; ++j) B[j] = load_frag_b(wb + wbase[j] + kb);
#pragma unroll
        for (int r = 0; r < RT; ++r)
#pragma unroll
            for (int j = 0; j < CW; ++j)
                acc[r][j] = __builtin_amdgcn_wmma_f32_16x16x32_f16(
                                false, A[r], false, B[j], (short)0,
                                acc[r][j], false, false);
    }
    __syncthreads();   // all reads of input activations complete

#pragma unroll
    for (int r = 0; r < RT; ++r) {
        const int row0 = (r << 4) + ((lane >> 4) << 3);  // C/D: lanes>=16 -> M+8
#pragma unroll
        for (int j = 0; j < CW; ++j) {
            int col  = (ctj[j] << 4) + (lane & 15);
            float bv = bbuf[L.offB + col];
#pragma unroll
            for (int q = 0; q < 8; ++q)
                acts[(row0 + q) * ST + col] = (_Float16)(acc[r][j][q] + bv);
        }
    }
    __syncthreads();
}

// ---------------------------------------------------------------------------
// Activation / tangent-mask pass over cols [0,256).
// mode 0: relu on rows 0..15 only (color net)
// mode 1: relu primal + (pre>0) mask on tangent rows (deform net)
// mode 2: softplus100 primal + sigmoid(100*pre) mask on tangent rows (sdf net)
__device__ void apply_act(_Float16* acts, int R, int mode, int tid)
{
    if (mode) {
        int n = (R - 16) * 256;
        for (int i = tid; i < n; i += 256) {
            int r = 16 + (i >> 8), c = i & 255;
            float t  = (float)acts[r * ST + c];
            float pp = (float)acts[(r & 15) * ST + c];  // primal pre-activation
            float dv = (mode == 1) ? (pp > 0.f ? 1.f : 0.f) : sigmoidf_(100.f * pp);
            acts[r * ST + c] = (_Float16)(t * dv);
        }
    }
    __syncthreads();
    for (int i = tid; i < 16 * 256; i += 256) {
        int r = i >> 8, c = i & 255;
        float pp = (float)acts[r * ST + c];
        float a  = (mode == 2) ? softplus100(pp) : fmaxf(pp, 0.f);
        acts[r * ST + c] = (_Float16)a;
    }
    __syncthreads();
}

// ---------------------------------------------------------------------------
// frequency encodings (one thread per point)
__device__ __forceinline__ void enc_primal(_Float16* row, const float* v, int n, int L)
{
    for (int i = 0; i < n; ++i) row[i] = (_Float16)v[i];
    for (int i = 0; i < n; ++i) {
        float f = 1.f;
        for (int k = 0; k < L; ++k) {
            float xb = v[i] * f;
            row[n + i * 2 * L + k]     = (_Float16)__sinf(xb);
            row[n + i * 2 * L + L + k] = (_Float16)__cosf(xb);
            f *= 2.f;
        }
    }
}
__device__ __forceinline__ void enc_tangent(_Float16* row, const float* v,
                                            const float* tv, int n, int L)
{
    for (int i = 0; i < n; ++i) row[i] = (_Float16)tv[i];
    for (int i = 0; i < n; ++i) {
        float f = 1.f;
        for (int k = 0; k < L; ++k) {
            float xb = v[i] * f;
            row[n + i * 2 * L + k]     = (_Float16)( __cosf(xb) * f * tv[i]);
            row[n + i * 2 * L + L + k] = (_Float16)(-__sinf(xb) * f * tv[i]);
            f *= 2.f;
        }
    }
}

// ---------------------------------------------------------------------------
// weight repack: f32 row-major [din][dout] -> blocked f16 WMMA B-fragments
// dst[(ct*ktiles+kt)*512 + lane*16 + j], lane<16: K j0..7->+0..7, j8..15->+16..23,
// lane>=16 shifts K by +8. Skip-layer 1/sqrt(2) folded via `scale`.
__global__ void endosurf_pack(const float* __restrict__ W, const float* __restrict__ b,
                              _Float16* __restrict__ dst, float* __restrict__ bdst,
                              int din, int dout, int ktiles, int ctiles, float scale)
{
    int total = ktiles * ctiles * 512;
    int end   = total + ctiles * 16;
    for (int i = blockIdx.x * blockDim.x + threadIdx.x; i < end;
         i += gridDim.x * blockDim.x) {
        if (i < total) {
            int j    = i & 15;
            int lane = (i >> 4) & 31;
            int tile = i >> 9;
            int kt   = tile % ktiles;
            int ct   = tile / ktiles;
            int c    = ct * 16 + (lane & 15);
            int koff = (j & 7) + ((j >> 3) << 4) + ((lane >> 4) << 3);
            int k    = kt * 32 + koff;
            float v  = (k < din && c < dout) ? W[k * dout + c] * scale : 0.f;
            dst[i] = (_Float16)v;
        } else {
            int c = i - total;
            bdst[c] = (c < dout) ? b[c] : 0.f;
        }
    }
}

// ---------------------------------------------------------------------------
__global__ __launch_bounds__(256)
void endosurf_main(const float* __restrict__ inp,
                   const _Float16* __restrict__ wbuf,
                   const float*    __restrict__ bbuf,
                   const float*    __restrict__ pvar,
                   float* __restrict__ out, AllDesc dd)
{
    __shared__ __align__(16) _Float16 acts[64 * ST];   // 45 KB: GEMM activations
    __shared__ __align__(16) _Float16 feat[16 * 256];  //  8 KB: geo features
    __shared__ float sx[16][3], sd[16][3], sxc[16][3], sdc[16][3], sg[16][3];
    __shared__ float ssdf[16];

    const int tid   = threadIdx.x;
    const int wave  = __builtin_amdgcn_readfirstlane(tid >> 5);
    const int lane  = tid & 31;
    const int gbase = blockIdx.x * 16;

    gh_t* wg = (gh_t*)wbuf;
    gf_t* bg = (gf_t*)bbuf;

    // ================= deform net: rows 0..15 primal, 16..31 tangent(d) ======
    for (int i = tid; i < 32 * ST; i += 256) acts[i] = (_Float16)0.f;
    __syncthreads();
    if (tid < 16) {
        int g = gbase + tid;
        float x[3], d[3];
        for (int i = 0; i < 3; ++i) {
            x[i] = inp[g * 7 + i];     sx[tid][i] = x[i];
            d[i] = inp[g * 7 + 3 + i]; sd[tid][i] = d[i];
        }
        float t = inp[g * 7 + 6];
        enc_primal(&acts[tid * ST], x, 3, 6);          // 39 vals
        enc_primal(&acts[tid * ST + 39], &t, 1, 6);    // 13 vals -> 52
        for (int i = 0; i < 52; ++i)                   // park for skip layer
            acts[tid * ST + 256 + i] = acts[tid * ST + i];
        enc_tangent(&acts[(16 + tid) * ST], x, d, 3, 6);  // t-part tangent = 0
        for (int i = 0; i < 52; ++i)
            acts[(16 + tid) * ST + 256 + i] = acts[(16 + tid) * ST + i];
    }
    __syncthreads();
    for (int l = 0; l < 8; ++l) {
        run_layer_t<2, 2>(acts, wg, bg, dd.L[l], wave, lane);
        apply_act(acts, 32, 1, tid);
    }
    run_layer_t<2, 1>(acts, wg, bg, dd.L[8], wave, lane);   // 256 -> 3 head
    if (tid < 16) {
        float nrm = 0.f;
        for (int i = 0; i < 3; ++i) {
            sxc[tid][i] = sx[tid][i] + (float)acts[tid * ST + i];
            float dc = sd[tid][i] + (float)acts[(16 + tid) * ST + i];
            sdc[tid][i] = dc; nrm += dc * dc;
        }
        nrm = sqrtf(nrm) + 1e-10f;
        for (int i = 0; i < 3; ++i) sdc[tid][i] /= nrm;
    }
    __syncthreads();

    // ====== sdf net: rows 0..15 primal, 16..63 = d/dx, d/dy, d/dz tangents ===
    for (int i = tid; i < 64 * ST; i += 256) acts[i] = (_Float16)0.f;
    __syncthreads();
    if (tid < 16) {
        float xc[3];
        for (int i = 0; i < 3; ++i) xc[i] = sxc[tid][i];
        enc_primal(&acts[tid * ST], xc, 3, 6);         // 39 vals
        for (int i = 0; i < 39; ++i)
            acts[tid * ST + 256 + i] = acts[tid * ST + i];
        for (int j = 0; j < 3; ++j) {
            float e[3] = {0.f, 0.f, 0.f}; e[j] = 1.f;
            _Float16* r = &acts[(16 * (1 + j) + tid) * ST];
            enc_tangent(r, xc, e, 3, 6);
            for (int i = 0; i < 39; ++i) r[256 + i] = r[i];
        }
    }
    __syncthreads();
    for (int l = 9; l < 17; ++l) {
        run_layer_t<4, 2>(acts, wg, bg, dd.L[l], wave, lane);
        apply_act(acts, 64, 2, tid);
    }
    run_layer_t<4, 3>(acts, wg, bg, dd.L[17], wave, lane);  // 256 -> 257 head
    if (tid < 16) {
        ssdf[tid] = (float)acts[tid * ST + 0];                       // sdf
        for (int j = 0; j < 3; ++j)
            sg[tid][j] = (float)acts[(16 * (1 + j) + tid) * ST + 0]; // gradient
    }
    for (int i = tid; i < 16 * 256; i += 256) {        // geo_feat = h[1:257]
        int r = i >> 8, c = i & 255;
        feat[i] = acts[r * ST + 1 + c];
    }
    __syncthreads();

    // ================= color net: 16 primal rows ============================
    for (int i = tid; i < 16 * ST; i += 256) acts[i] = (_Float16)0.f;
    __syncthreads();
    if (tid < 16) {
        enc_primal(&acts[tid * ST], sxc[tid], 3, 10);                 // 63
        for (int i = 0; i < 3; ++i)
            acts[tid * ST + 63 + i] = (_Float16)sg[tid][i];           // n
        enc_primal(&acts[tid * ST + 66], sdc[tid], 3, 4);             // 27 -> 93
    }
    for (int i = tid; i < 16 * 256; i += 256) {
        int r = i >> 8, c = i & 255;
        acts[r * ST + 93 + c] = feat[i];                              // 93..348
    }
    __syncthreads();
    for (int l = 18; l < 22; ++l) {
        run_layer_t<1, 2>(acts, wg, bg, dd.L[l], wave, lane);
        apply_act(acts, 16, 0, tid);
    }
    run_layer_t<1, 1>(acts, wg, bg, dd.L[22], wave, lane);  // 256 -> 3 head

    // ================= output: [color3, sdf, inv_s, grad3] ==================
    if (tid < 16) {
        int g = gbase + tid;
        float inv_s = __expf(10.f * pvar[0]);
        for (int i = 0; i < 3; ++i)
            out[g * 8 + i] = sigmoidf_((float)acts[tid * ST + i]);
        out[g * 8 + 3] = ssdf[tid];
        out[g * 8 + 4] = inv_s;
        for (int i = 0; i < 3; ++i) out[g * 8 + 5 + i] = sg[tid][i];
    }
}

// ---------------------------------------------------------------------------
extern "C" void kernel_launch(void* const* d_in, const int* in_sizes, int n_in,
                              void* d_out, int out_size, void* d_ws, size_t ws_size,
                              hipStream_t stream)
{
    static const int din[NLAYERS] = {
        52, 256, 256, 256, 256, 308, 256, 256, 256,   // deform (skip @5)
        39, 256, 256, 256, 256, 295, 256, 256, 256,   // sdf    (skip @5)
        349, 256, 256, 256, 256                        // color
    };
    static const int dout[NLAYERS] = {
        256, 256, 256, 256, 256, 256, 256, 256, 3,
        256, 256, 256, 256, 256, 256, 256, 256, 257,
        256, 256, 256, 256, 3
    };
    const float INVS2 = 0.70710678118654752f;

    AllDesc dd;
    size_t woff = 0, boff = 0;
    for (int l = 0; l < NLAYERS; ++l) {
        int kt = (din[l] + 31) / 32;
        int ct = (dout[l] + 15) / 16;
        dd.L[l].offW = (int)woff; dd.L[l].offB = (int)boff;
        dd.L[l].ktiles = kt;      dd.L[l].ctiles = ct;
        woff += (size_t)kt * ct * 512;   // halfs
        boff += (size_t)ct * 16;         // floats
    }
    _Float16* wbuf = (_Float16*)d_ws;
    float*    bbuf = (float*)((char*)d_ws + woff * sizeof(_Float16));

    for (int l = 0; l < NLAYERS; ++l) {
        float scale = (l == 5 || l == 14) ? INVS2 : 1.f;   // fold concat/sqrt(2)
        int total  = dd.L[l].ktiles * dd.L[l].ctiles * 512 + dd.L[l].ctiles * 16;
        int blocks = (total + 255) / 256;
        endosurf_pack<<<blocks, 256, 0, stream>>>(
            (const float*)d_in[1 + 2 * l], (const float*)d_in[2 + 2 * l],
            wbuf + dd.L[l].offW, bbuf + dd.L[l].offB,
            din[l], dout[l], dd.L[l].ktiles, dd.L[l].ctiles, scale);
    }

    const int N = 65536;
    endosurf_main<<<N / 16, 256, 0, stream>>>(
        (const float*)d_in[0], wbuf, bbuf,
        (const float*)d_in[n_in - 1], (float*)d_out, dd);
}
